// subLSTMCell_46883863003217
// MI455X (gfx1250) — matrix operand order repaired and checked by reading, not compiled, not run on hardware
//
#include <hip/hip_runtime.h>

// ---------------------------------------------------------------------------
// subLSTM cell + Dense classifier for MI455X (gfx1250), bf16 WMMA path.
//   z = sigmoid(X@W + b + H@R); i,f,cg,o = split(z); c' = f*c + cg - i
//   h' = sigmoid(c') - o; out = h'@Wc + bc
// ~309 GFLOP vs ~420MB HBM traffic (~735 flop/byte) -> compute bound.
// v_wmma_f32_16x16x32_bf16 with f32 accumulate; double-buffered LDS with
// register staging (1 barrier / K-step); paired-row B staging so transposed
// LDS stores are dword-wide.
// ---------------------------------------------------------------------------

typedef __attribute__((ext_vector_type(16))) __bf16 v16bf;
typedef __attribute__((ext_vector_type(8)))  __bf16 v8bf;
typedef __attribute__((ext_vector_type(4)))  __bf16 v4bf;
typedef __attribute__((ext_vector_type(2)))  __bf16 v2bf;
typedef __attribute__((ext_vector_type(8)))  float  v8f;

#define B_DIM 16384
#define D_IN  1024
#define UNITS 1024
#define D_OUT 1024

__device__ __forceinline__ float sigm(float x) {
    return __builtin_amdgcn_rcpf(1.0f + __expf(-x));   // v_exp + v_rcp
}
__device__ __forceinline__ __bf16 bf(float x) { return (__bf16)x; }
__device__ __forceinline__ void stpk(__bf16* dst, float lo, float hi) {
    v2bf p; p[0] = bf(lo); p[1] = bf(hi);
    *(v2bf*)dst = p;                                    // one ds_store_b32
}

// Tiling (wave32, 256 threads = 8 waves per block)
constexpr int KB   = 32;   // WMMA K per step
constexpr int ASTR = 40;   // LDS row stride (elements), padded vs 32
constexpr int BSTR = 40;

// ============================ Kernel A: gates ==============================
// Per block: rows [rowBase, rowBase+128), gate-cols [nBase, nBase+32) for ALL
// 4 gates. K loop runs over concatenated K=2048: [X|H] @ [W;R].
// Waves 4(M)x2(N); each wave: 2 M-subtiles x 4 gates = 8 WMMA / K-step.
constexpr int BM_A = 128, BN_A = 32;

__global__ __launch_bounds__(256) void sublstm_gates_wmma(
    const float* __restrict__ X, const float* __restrict__ H,
    const float* __restrict__ Cin, const float* __restrict__ W,
    const float* __restrict__ R, const float* __restrict__ bias,
    float* __restrict__ Hn, float* __restrict__ Cn)
{
    __shared__ __align__(16) __bf16 As[2][BM_A * ASTR];       // row-major M x K
    __shared__ __align__(16) __bf16 Bs[2][4][BN_A * BSTR];    // transposed N x K

    const int tid   = threadIdx.x;
    const int wave  = tid >> 5;
    const int lane  = tid & 31;
    const int lhalf = lane >> 4;    // 0/1 -> K-half (A/B) / M-half (C) per ISA
    const int lmod  = lane & 15;
    const int waveM = wave >> 1;    // 0..3 -> 32 rows each
    const int waveN = wave & 1;     // 0..1 -> 16 gate-cols each
    const int rowBase = blockIdx.y * BM_A;
    const int nBase   = blockIdx.x * BN_A;

    v8f acc[2][4];
    #pragma unroll
    for (int mi = 0; mi < 2; ++mi)
        #pragma unroll
        for (int g = 0; g < 4; ++g)
            #pragma unroll
            for (int e = 0; e < 8; ++e) acc[mi][g][e] = 0.0f;

    float4 aR[4], bR0[2], bR1[2];   // register staging for next tile

    // global -> registers for K-step kt (A: 128x32, B: 4 x 32x32, row pairs)
    auto ldTile = [&](int kt) {
        const bool  ph    = (kt < D_IN);
        const float* aSrc = ph ? X : H;
        const float* bSrc = ph ? W : R;
        const int    ak   = ph ? kt : kt - D_IN;
        #pragma unroll
        for (int i = 0; i < 4; ++i) {
            int idx = tid + i * 256, r = idx >> 3, c4 = idx & 7;
            aR[i] = *(const float4*)(aSrc + (size_t)(rowBase + r) * D_IN + ak + c4 * 4);
        }
        #pragma unroll
        for (int i = 0; i < 2; ++i) {
            int idx = tid + i * 256;
            int g = idx >> 7, rem = idx & 127, kk2 = rem >> 3, c4 = rem & 7;
            const float* bp = bSrc + (size_t)(ak + 2 * kk2) * (4 * UNITS)
                              + g * UNITS + nBase + c4 * 4;
            bR0[i] = *(const float4*)(bp);
            bR1[i] = *(const float4*)(bp + 4 * UNITS);
        }
    };
    // registers -> bf16 LDS buffer p (B transposed, dword-packed K-pairs)
    auto stTile = [&](int p) {
        #pragma unroll
        for (int i = 0; i < 4; ++i) {
            int idx = tid + i * 256, r = idx >> 3, c4 = idx & 7;
            v4bf q; q[0] = bf(aR[i].x); q[1] = bf(aR[i].y);
                    q[2] = bf(aR[i].z); q[3] = bf(aR[i].w);
            *(v4bf*)(&As[p][r * ASTR + c4 * 4]) = q;
        }
        #pragma unroll
        for (int i = 0; i < 2; ++i) {
            int idx = tid + i * 256;
            int g = idx >> 7, rem = idx & 127, kk2 = rem >> 3, c4 = rem & 7;
            __bf16* d = &Bs[p][g][(c4 * 4) * BSTR + 2 * kk2];
            stpk(d + 0 * BSTR, bR0[i].x, bR1[i].x);
            stpk(d + 1 * BSTR, bR0[i].y, bR1[i].y);
            stpk(d + 2 * BSTR, bR0[i].z, bR1[i].z);
            stpk(d + 3 * BSTR, bR0[i].w, bR1[i].w);
        }
    };

    ldTile(0); stTile(0); __syncthreads();

    int p = 0;
    for (int kt = 0; kt < 2 * D_IN; kt += KB, p ^= 1) {
        const bool hasNext = (kt + KB) < 2 * D_IN;
        if (hasNext) ldTile(kt + KB);            // global loads overlap WMMAs

        v16bf afrag[2];
        #pragma unroll
        for (int mi = 0; mi < 2; ++mi) {
            const __bf16* ap = &As[p][(waveM * 32 + mi * 16 + lmod) * ASTR + lhalf * 8];
            ((v8bf*)&afrag[mi])[0] = *(const v8bf*)(ap);
            ((v8bf*)&afrag[mi])[1] = *(const v8bf*)(ap + 16);
        }
        #pragma unroll
        for (int g = 0; g < 4; ++g) {
            v16bf bfrag;
            const __bf16* bp = &Bs[p][g][(waveN * 16 + lmod) * BSTR + lhalf * 8];
            ((v8bf*)&bfrag)[0] = *(const v8bf*)(bp);
            ((v8bf*)&bfrag)[1] = *(const v8bf*)(bp + 16);
            #pragma unroll
            for (int mi = 0; mi < 2; ++mi)
                acc[mi][g] = __builtin_amdgcn_wmma_f32_16x16x32_bf16(
                    false, afrag[mi], false, bfrag, (short)0, acc[mi][g], false, false);
        }
        if (hasNext) stTile(p ^ 1);              // fill the other buffer
        __syncthreads();
    }

    // --- fused subtractive-gating epilogue (C/D layout: M=r+lhalf*8, N=lmod)
    const int col = nBase + waveN * 16 + lmod;   // u in [0, UNITS)
    const float b0 = bias[0 * UNITS + col];
    const float b1 = bias[1 * UNITS + col];
    const float b2 = bias[2 * UNITS + col];
    const float b3 = bias[3 * UNITS + col];
    #pragma unroll
    for (int mi = 0; mi < 2; ++mi)
        #pragma unroll
        for (int r = 0; r < 8; ++r) {
            const int row = rowBase + waveM * 32 + mi * 16 + lhalf * 8 + r;
            const float gi = sigm(acc[mi][0][r] + b0);
            const float gf = sigm(acc[mi][1][r] + b1);
            const float gc = sigm(acc[mi][2][r] + b2);
            const float go = sigm(acc[mi][3][r] + b3);
            const float cprev = Cin[(size_t)row * UNITS + col];
            const float cn = gf * cprev + gc - gi;
            const float hn = sigm(cn) - go;
            Hn[(size_t)row * UNITS + col] = hn;
            Cn[(size_t)row * UNITS + col] = cn;
        }
}

// ========================= Kernel B: classifier ============================
// out = h_new @ Wc + bc.  128x128 block tile, waves 4(M)x2(N); each wave a
// 32x64 macro tile (2x4 WMMA tiles) = 8 WMMA / K-step.
constexpr int BM_B = 128, BN_B = 128;

__global__ __launch_bounds__(256) void sublstm_dense_wmma(
    const float* __restrict__ Hn, const float* __restrict__ Wc,
    const float* __restrict__ bc, float* __restrict__ Out)
{
    __shared__ __align__(16) __bf16 As[2][BM_B * ASTR];
    __shared__ __align__(16) __bf16 Bs[2][BN_B * BSTR];       // transposed N x K

    const int tid   = threadIdx.x;
    const int wave  = tid >> 5;
    const int lane  = tid & 31;
    const int lhalf = lane >> 4;
    const int lmod  = lane & 15;
    const int waveM = wave >> 1;    // 0..3 -> 32 rows each
    const int waveN = wave & 1;     // 0..1 -> 64 cols each
    const int rowBase = blockIdx.y * BM_B;
    const int nBase   = blockIdx.x * BN_B;

    v8f acc[2][4];
    #pragma unroll
    for (int mi = 0; mi < 2; ++mi)
        #pragma unroll
        for (int ni = 0; ni < 4; ++ni)
            #pragma unroll
            for (int e = 0; e < 8; ++e) acc[mi][ni][e] = 0.0f;

    float4 aR[4], bR0[2], bR1[2];

    auto ldTile = [&](int kt) {
        #pragma unroll
        for (int i = 0; i < 4; ++i) {
            int idx = tid + i * 256, r = idx >> 3, c4 = idx & 7;
            aR[i] = *(const float4*)(Hn + (size_t)(rowBase + r) * UNITS + kt + c4 * 4);
        }
        #pragma unroll
        for (int i = 0; i < 2; ++i) {
            int idx = tid + i * 256, kk2 = idx >> 5, c4 = idx & 31;
            const float* bp = Wc + (size_t)(kt + 2 * kk2) * D_OUT + nBase + c4 * 4;
            bR0[i] = *(const float4*)(bp);
            bR1[i] = *(const float4*)(bp + D_OUT);
        }
    };
    auto stTile = [&](int p) {
        #pragma unroll
        for (int i = 0; i < 4; ++i) {
            int idx = tid + i * 256, r = idx >> 3, c4 = idx & 7;
            v4bf q; q[0] = bf(aR[i].x); q[1] = bf(aR[i].y);
                    q[2] = bf(aR[i].z); q[3] = bf(aR[i].w);
            *(v4bf*)(&As[p][r * ASTR + c4 * 4]) = q;
        }
        #pragma unroll
        for (int i = 0; i < 2; ++i) {
            int idx = tid + i * 256, kk2 = idx >> 5, c4 = idx & 31;
            __bf16* d = &Bs[p][(c4 * 4) * BSTR + 2 * kk2];
            stpk(d + 0 * BSTR, bR0[i].x, bR1[i].x);
            stpk(d + 1 * BSTR, bR0[i].y, bR1[i].y);
            stpk(d + 2 * BSTR, bR0[i].z, bR1[i].z);
            stpk(d + 3 * BSTR, bR0[i].w, bR1[i].w);
        }
    };

    ldTile(0); stTile(0); __syncthreads();

    int p = 0;
    for (int kt = 0; kt < UNITS; kt += KB, p ^= 1) {
        const bool hasNext = (kt + KB) < UNITS;
        if (hasNext) ldTile(kt + KB);

        v16bf afrag[2];
        #pragma unroll
        for (int mi = 0; mi < 2; ++mi) {
            const __bf16* ap = &As[p][(waveM * 32 + mi * 16 + lmod) * ASTR + lhalf * 8];
            ((v8bf*)&afrag[mi])[0] = *(const v8bf*)(ap);
            ((v8bf*)&afrag[mi])[1] = *(const v8bf*)(ap + 16);
        }
        #pragma unroll
        for (int ni = 0; ni < 4; ++ni) {
            v16bf bfrag;
            const __bf16* bp = &Bs[p][(waveN * 64 + ni * 16 + lmod) * BSTR + lhalf * 8];
            ((v8bf*)&bfrag)[0] = *(const v8bf*)(bp);
            ((v8bf*)&bfrag)[1] = *(const v8bf*)(bp + 16);
            #pragma unroll
            for (int mi = 0; mi < 2; ++mi)
                acc[mi][ni] = __builtin_amdgcn_wmma_f32_16x16x32_bf16(
                    false, afrag[mi], false, bfrag, (short)0, acc[mi][ni], false, false);
        }
        if (hasNext) stTile(p ^ 1);
        __syncthreads();
    }

    #pragma unroll
    for (int ni = 0; ni < 4; ++ni) {
        const int col = nBase + waveN * 64 + ni * 16 + lmod;
        const float bcol = bc[col];
        #pragma unroll
        for (int mi = 0; mi < 2; ++mi)
            #pragma unroll
            for (int r = 0; r < 8; ++r) {
                const int row = rowBase + waveM * 32 + mi * 16 + lhalf * 8 + r;
                Out[(size_t)row * D_OUT + col] = acc[mi][ni][r] + bcol;
            }
    }
}

// ================================ launch ===================================
extern "C" void kernel_launch(void* const* d_in, const int* in_sizes, int n_in,
                              void* d_out, int out_size, void* d_ws, size_t ws_size,
                              hipStream_t stream) {
    (void)in_sizes; (void)n_in; (void)out_size; (void)d_ws; (void)ws_size;

    const float* inputs = (const float*)d_in[0];
    const float* h_tm1  = (const float*)d_in[1];
    const float* c_tm1  = (const float*)d_in[2];
    const float* kern   = (const float*)d_in[3];
    const float* rkern  = (const float*)d_in[4];
    const float* bias   = (const float*)d_in[5];
    const float* Wc     = (const float*)d_in[6];
    const float* bc     = (const float*)d_in[7];

    float* out   = (float*)d_out;                          // [B, D_OUT]
    float* h_new = out   + (size_t)B_DIM * UNITS;          // [B, UNITS]
    float* c_new = h_new + (size_t)B_DIM * UNITS;          // [B, UNITS]

    dim3 gA(UNITS / BN_A, B_DIM / BM_A);   // (32, 128)
    sublstm_gates_wmma<<<gA, 256, 0, stream>>>(inputs, h_tm1, c_tm1, kern, rkern,
                                               bias, h_new, c_new);

    dim3 gB(D_OUT / BN_B, B_DIM / BM_B);   // (8, 128)
    sublstm_dense_wmma<<<gB, 256, 0, stream>>>(h_new, Wc, bc, out);
}
